// SelfAttention_34583076668084
// MI455X (gfx1250) — matrix operand order repaired and checked by reading, not compiled, not run on hardware
//
#include <hip/hip_runtime.h>
#include <hip/hip_bf16.h>

// ---------------------------------------------------------------------------
// Self-attention forward for gfx1250 (MI455X), f16 WMMA with fp32 accumulate.
// B=1, S=4096, D=1024, H=16, HD=64. Flash-attention style (no score matrix).
// ---------------------------------------------------------------------------

typedef __attribute__((ext_vector_type(16))) _Float16 v16h;
typedef __attribute__((ext_vector_type(8)))  _Float16 v8h;
typedef __attribute__((ext_vector_type(8)))  float    v8f;

union V16U { v16h v; v8h h[2]; };

#define S_LEN 4096
#define DMODEL 1024
#define NHEAD 16
#define HDIM 64

// A-fragment (16x32 f16): lane l<16 holds row l, K = {0..7, 16..23};
// lane l>=16 holds row l-16, K = {8..15, 24..31}.  (CDNA5 ISA 7.12.2)
__device__ __forceinline__ v16h load_fragA(const _Float16* p, int stride, int kbase) {
  const int l = threadIdx.x & 31;
  const _Float16* r = p + (size_t)(l & 15) * stride + kbase + ((l >> 4) << 3);
  V16U u;
  u.h[0] = *(const v8h*)(r);
  u.h[1] = *(const v8h*)(r + 16);
  return u.v;
}

// B-fragment (32x16 f16) from N-major storage: lane l<16 holds col l, K = {0..7, 16..23};
// lane l>=16 holds col l-16, K = {8..15, 24..31}.
__device__ __forceinline__ v16h load_fragB(const _Float16* p, int stride, int kbase) {
  const int l = threadIdx.x & 31;
  const _Float16* r = p + (size_t)(l & 15) * stride + kbase + ((l >> 4) << 3);
  V16U u;
  u.h[0] = *(const v8h*)(r);
  u.h[1] = *(const v8h*)(r + 16);
  return u.v;
}

// 32-byte (16 x f16) contiguous load as two b128s
__device__ __forceinline__ V16U load_32B(const _Float16* p) {
  V16U u;
  u.h[0] = *(const v8h*)(p);
  u.h[1] = *(const v8h*)(p + 8);
  return u;
}

__device__ __forceinline__ v8f wmma_f16(v16h a, v16h b, v8f c) {
  return __builtin_amdgcn_wmma_f32_16x16x32_f16(false, a, false, b, (short)0, c, false, false);
}

// ---------------------------------------------------------------------------
// Conversion kernels
// ---------------------------------------------------------------------------
__global__ void cvt_f32_to_f16_kernel(const float* __restrict__ src,
                                      _Float16* __restrict__ dst, int n) {
  int i = blockIdx.x * blockDim.x + threadIdx.x;
  if (i < n) dst[i] = (_Float16)src[i];
}

// dst[c*rows + r] = (f16)src[r*cols + c]
__global__ void transpose_cvt_kernel(const float* __restrict__ src,
                                     _Float16* __restrict__ dst,
                                     int rows, int cols) {
  int i = blockIdx.x * blockDim.x + threadIdx.x;
  if (i < rows * cols) {
    int r = i / cols, c = i % cols;
    dst[(size_t)c * rows + r] = (_Float16)src[i];
  }
}

// ---------------------------------------------------------------------------
// QKV projection (64x64 tile / wave) + RoPE epilogue for Q,K.
// Q,K stored row-major [s][1024] f16; V stored transposed [h*64+e][s] f16.
// ---------------------------------------------------------------------------
__global__ void __launch_bounds__(128)
qkv_rope_kernel(const _Float16* __restrict__ xh,
                const _Float16* __restrict__ wqt,
                const _Float16* __restrict__ wkt,
                const _Float16* __restrict__ wvt,
                _Float16* __restrict__ qf,
                _Float16* __restrict__ kf,
                _Float16* __restrict__ vt) {
  const int lane = threadIdx.x & 31;
  const int wave = blockIdx.x * (blockDim.x >> 5) + (threadIdx.x >> 5);
  const int w   = wave / (64 * NHEAD);          // 0=Q, 1=K, 2=V
  const int rem = wave % (64 * NHEAD);
  const int m0  = (rem >> 4) * 64;              // sequence tile
  const int h   = rem & 15;                     // head == 64-wide N tile
  const int n0  = h * HDIM;
  const _Float16* wt = (w == 0) ? wqt : (w == 1) ? wkt : wvt;

  v8f acc[4][4] = {};

  for (int k = 0; k < DMODEL; k += 32) {
    v16h a[4], b[4];
#pragma unroll
    for (int i = 0; i < 4; ++i)
      a[i] = load_fragA(xh + (size_t)(m0 + i * 16) * DMODEL, DMODEL, k);
#pragma unroll
    for (int j = 0; j < 4; ++j)
      b[j] = load_fragB(wt + (size_t)(n0 + j * 16) * DMODEL, DMODEL, k);
    if (k + 32 < DMODEL) {  // WGP-scope stream hint -> global_prefetch_b8
      __builtin_prefetch(xh + (size_t)m0 * DMODEL + k + 32, 0, 3);
      __builtin_prefetch(wt + (size_t)n0 * DMODEL + k + 32, 0, 3);
    }
#pragma unroll
    for (int i = 0; i < 4; ++i)
#pragma unroll
      for (int j = 0; j < 4; ++j)
        acc[i][j] = wmma_f16(a[i], b[j], acc[i][j]);
  }

  const int hi = lane >> 4;
  const int cc = lane & 15;

  if (w < 2) {
    // RoPE: q'[e] = q[e]cos - q[e+32]sin ; q'[e+32] = q[e+32]cos + q[e]sin
    // pair lives in fragments (ns, ns+2), same lane / same j.
#pragma unroll
    for (int i = 0; i < 4; ++i) {
#pragma unroll
      for (int ns = 0; ns < 2; ++ns) {
        const int e = ns * 16 + cc;  // 0..31
        // freq = 10000^(-e/32) = exp2(-e * log2(10000)/32)
        const float fr = exp2f(-0.41524101186092204f * (float)e);
#pragma unroll
        for (int j = 0; j < 8; ++j) {
          const int srow = m0 + i * 16 + j + 8 * hi;
          float sn, cs;
          __sincosf((float)srow * fr, &sn, &cs);
          const float lo = acc[i][ns][j];
          const float hv = acc[i][ns + 2][j];
          acc[i][ns][j]     = lo * cs - hv * sn;
          acc[i][ns + 2][j] = hv * cs + lo * sn;
        }
      }
    }
    _Float16* dst = (w == 0) ? qf : kf;
#pragma unroll
    for (int i = 0; i < 4; ++i)
#pragma unroll
      for (int ns = 0; ns < 4; ++ns)
#pragma unroll
        for (int j = 0; j < 8; ++j)
          dst[(size_t)(m0 + i * 16 + j + 8 * hi) * DMODEL + n0 + ns * 16 + cc] =
              (_Float16)acc[i][ns][j];
  } else {
    // V transposed: vt[(h*64+e)][s]
#pragma unroll
    for (int i = 0; i < 4; ++i)
#pragma unroll
      for (int ns = 0; ns < 4; ++ns)
#pragma unroll
        for (int j = 0; j < 8; ++j)
          vt[(size_t)(n0 + ns * 16 + cc) * S_LEN + m0 + i * 16 + j + 8 * hi] =
              (_Float16)acc[i][ns][j];
  }
}

// ---------------------------------------------------------------------------
// Flash attention, block-cooperative: 4 waves = one 64-row q block of one
// head; K/V tiles double-buffered in LDS (8 KB/iter shared by all 4 waves).
// ---------------------------------------------------------------------------
__global__ void __launch_bounds__(128)
flash_attn_kernel(const _Float16* __restrict__ qf,
                  const _Float16* __restrict__ kf,
                  const _Float16* __restrict__ vt,
                  _Float16* __restrict__ cx) {
  __shared__ _Float16 ldsK[2][32 * 64];  // [buf][k_local][e]
  __shared__ _Float16 ldsV[2][64 * 32];  // [buf][e][k_local]
  __shared__ _Float16 ldsP[4][16 * 32];  // per-wave P staging

  const int tid  = threadIdx.x;
  const int lane = tid & 31;
  const int wv   = tid >> 5;
  const int h    = blockIdx.x >> 6;       // 64 q-blocks per head
  const int qb   = (blockIdx.x & 63) * 64;
  const int q0   = qb + wv * 16;
  const int hi = lane >> 4, cc = lane & 15;

  const _Float16* qh = qf + h * HDIM;
  const _Float16* kh = kf + h * HDIM;                   // [s][1024] (head cols)
  const _Float16* vh = vt + (size_t)h * HDIM * S_LEN;   // [e][s]

  // cooperative copy: each of 128 threads moves one 32B chunk per tile
  const int krow = tid >> 2, kcol = (tid & 3) << 4;     // K: 32 rows x 64 e
  const int vrow = tid >> 1, vcol = (tid & 1) << 4;     // V: 64 rows x 32 s

  const v16h aq0 = load_fragA(qh + (size_t)q0 * DMODEL, DMODEL, 0);
  const v16h aq1 = load_fragA(qh + (size_t)q0 * DMODEL, DMODEL, 32);

  v8f accv[4] = {};
  float mrow[8], lrow[8], alpha[8];
#pragma unroll
  for (int j = 0; j < 8; ++j) { mrow[j] = -1e30f; lrow[j] = 0.0f; }

  const int kend = qb + 64;  // causal bound for the whole 64-row q block
  _Float16* lp = ldsP[wv];

  // prologue: first K/V tile into registers
  V16U kreg = load_32B(kh + (size_t)krow * DMODEL + kcol);
  V16U vreg = load_32B(vh + (size_t)vrow * S_LEN + vcol);

  int buf = 0;
  for (int kb = 0; kb < kend; kb += 32) {
    // stage current tile into LDS
    *(v8h*)&ldsK[buf][krow * 64 + kcol]     = kreg.h[0];
    *(v8h*)&ldsK[buf][krow * 64 + kcol + 8] = kreg.h[1];
    *(v8h*)&ldsV[buf][vrow * 32 + vcol]     = vreg.h[0];
    *(v8h*)&ldsV[buf][vrow * 32 + vcol + 8] = vreg.h[1];
    // issue next tile's global loads early (overlap with compute)
    if (kb + 32 < kend) {
      kreg = load_32B(kh + (size_t)(kb + 32 + krow) * DMODEL + kcol);
      vreg = load_32B(vh + (size_t)vrow * S_LEN + kb + 32 + vcol);
    }
    __syncthreads();

    // ---- scores: S[16 x 32] = Q (16x64) * K^T (K-dim = head dim, 2 WMMAs)
    v8f s0 = {}, s1 = {};
    {
      const _Float16* kt = ldsK[buf];
      v16h b0 = load_fragB(kt, 64, 0);
      v16h b1 = load_fragB(kt, 64, 32);
      s0 = wmma_f16(aq0, b0, s0);
      s0 = wmma_f16(aq1, b1, s0);
      v16h b2 = load_fragB(kt + 16 * 64, 64, 0);
      v16h b3 = load_fragB(kt + 16 * 64, 64, 32);
      s1 = wmma_f16(aq0, b2, s1);
      s1 = wmma_f16(aq1, b3, s1);
    }

    const bool needMask = (kb + 31 > q0);
#pragma unroll
    for (int j = 0; j < 8; ++j) {
      const int qi = q0 + j + 8 * hi;
      float v0 = s0[j] * 0.125f;  // 1/sqrt(64)
      float v1 = s1[j] * 0.125f;
      if (needMask) {
        if (kb + cc      > qi) v0 = -1e30f;
        if (kb + 16 + cc > qi) v1 = -1e30f;
      }
      // row max across the 16 lanes holding this row (xor stays in group)
      float mx = fmaxf(v0, v1);
      mx = fmaxf(mx, __shfl_xor(mx, 1));
      mx = fmaxf(mx, __shfl_xor(mx, 2));
      mx = fmaxf(mx, __shfl_xor(mx, 4));
      mx = fmaxf(mx, __shfl_xor(mx, 8));
      const float mn = fmaxf(mrow[j], mx);
      alpha[j] = __expf(mrow[j] - mn);
      mrow[j]  = mn;
      v0 = __expf(v0 - mn);
      v1 = __expf(v1 - mn);
      float rs = v0 + v1;
      rs += __shfl_xor(rs, 1);
      rs += __shfl_xor(rs, 2);
      rs += __shfl_xor(rs, 4);
      rs += __shfl_xor(rs, 8);
      lrow[j] = lrow[j] * alpha[j] + rs;
      // stage P (C layout) into per-wave LDS for A-fragment reshape
      const int r = j + 8 * hi;
      lp[r * 32 + cc]      = (_Float16)v0;
      lp[r * 32 + 16 + cc] = (_Float16)v1;
    }
#pragma unroll
    for (int et = 0; et < 4; ++et)
#pragma unroll
      for (int j = 0; j < 8; ++j) accv[et][j] *= alpha[j];

    asm volatile("" ::: "memory");  // keep ds stores before ds loads (in-order HW)
    const v16h pf = load_fragA(lp, 32, 0);

    // ---- ctx += P (16x32) * V (32x64): 4 WMMAs over head-dim tiles
#pragma unroll
    for (int et = 0; et < 4; ++et) {
      const v16h bv = load_fragB(&ldsV[buf][et * 16 * 32], 32, 0);
      accv[et] = wmma_f16(pf, bv, accv[et]);
    }
    buf ^= 1;
  }

  float inv[8];
#pragma unroll
  for (int j = 0; j < 8; ++j) inv[j] = 1.0f / lrow[j];
#pragma unroll
  for (int et = 0; et < 4; ++et)
#pragma unroll
    for (int j = 0; j < 8; ++j)
      cx[(size_t)(q0 + j + 8 * hi) * DMODEL + h * HDIM + et * 16 + cc] =
          (_Float16)(accv[et][j] * inv[j]);
}

// ---------------------------------------------------------------------------
// Output projection: out[s][d] = ctx[s][he] * WoT[d][he], fp32 output.
// ---------------------------------------------------------------------------
__global__ void __launch_bounds__(128)
out_proj_kernel(const _Float16* __restrict__ ch,
                const _Float16* __restrict__ wot,
                float* __restrict__ out) {
  const int lane = threadIdx.x & 31;
  const int wave = blockIdx.x * (blockDim.x >> 5) + (threadIdx.x >> 5);
  const int m0 = (wave >> 4) * 64;
  const int n0 = (wave & 15) * 64;

  v8f acc[4][4] = {};
  for (int k = 0; k < DMODEL; k += 32) {
    v16h a[4], b[4];
#pragma unroll
    for (int i = 0; i < 4; ++i)
      a[i] = load_fragA(ch + (size_t)(m0 + i * 16) * DMODEL, DMODEL, k);
#pragma unroll
    for (int j = 0; j < 4; ++j)
      b[j] = load_fragB(wot + (size_t)(n0 + j * 16) * DMODEL, DMODEL, k);
    if (k + 32 < DMODEL) {
      __builtin_prefetch(ch + (size_t)m0 * DMODEL + k + 32, 0, 3);
      __builtin_prefetch(wot + (size_t)n0 * DMODEL + k + 32, 0, 3);
    }
#pragma unroll
    for (int i = 0; i < 4; ++i)
#pragma unroll
      for (int j = 0; j < 4; ++j)
        acc[i][j] = wmma_f16(a[i], b[j], acc[i][j]);
  }
  const int hi = lane >> 4, cc = lane & 15;
#pragma unroll
  for (int i = 0; i < 4; ++i)
#pragma unroll
    for (int ns = 0; ns < 4; ++ns)
#pragma unroll
      for (int j = 0; j < 8; ++j)
        out[(size_t)(m0 + i * 16 + j + 8 * hi) * DMODEL + n0 + ns * 16 + cc] =
            acc[i][ns][j];
}

// ---------------------------------------------------------------------------
// Host launcher
// ---------------------------------------------------------------------------
extern "C" void kernel_launch(void* const* d_in, const int* in_sizes, int n_in,
                              void* d_out, int out_size, void* d_ws, size_t ws_size,
                              hipStream_t stream) {
  (void)in_sizes; (void)n_in; (void)out_size; (void)ws_size;
  const float* x  = (const float*)d_in[0];
  // d_in[1] is the causal mask (implicit in the kernel)
  const float* Wq = (const float*)d_in[2];
  const float* Wk = (const float*)d_in[3];
  const float* Wv = (const float*)d_in[4];
  const float* Wo = (const float*)d_in[5];
  float* out = (float*)d_out;

  char* ws = (char*)d_ws;
  const size_t MB = (size_t)1 << 20;
  _Float16* xh  = (_Float16*)(ws);             //  8 MB  x f16 [4096][1024]
  _Float16* wqt = (_Float16*)(ws +  8 * MB);   //  2 MB  Wq^T [n][k]
  _Float16* wkt = (_Float16*)(ws + 10 * MB);   //  2 MB
  _Float16* wvt = (_Float16*)(ws + 12 * MB);   //  2 MB
  _Float16* wot = (_Float16*)(ws + 14 * MB);   //  2 MB  Wo^T [d][he]
  _Float16* qf  = (_Float16*)(ws + 16 * MB);   //  8 MB  Q (rope'd) [s][1024]
  _Float16* kf  = (_Float16*)(ws + 24 * MB);   //  8 MB  K (rope'd) [s][1024]
  _Float16* vt  = (_Float16*)(ws + 32 * MB);   //  8 MB  V^T [he][s]
  _Float16* cx  = (_Float16*)(ws + 40 * MB);   //  8 MB  ctx [s][1024]

  // 1) precision conversion / weight transposition (one-time, L2-resident)
  cvt_f32_to_f16_kernel<<<(S_LEN * DMODEL) / 256, 256, 0, stream>>>(x, xh, S_LEN * DMODEL);
  transpose_cvt_kernel<<<(DMODEL * DMODEL) / 256, 256, 0, stream>>>(Wq, wqt, DMODEL, DMODEL);
  transpose_cvt_kernel<<<(DMODEL * DMODEL) / 256, 256, 0, stream>>>(Wk, wkt, DMODEL, DMODEL);
  transpose_cvt_kernel<<<(DMODEL * DMODEL) / 256, 256, 0, stream>>>(Wv, wvt, DMODEL, DMODEL);
  transpose_cvt_kernel<<<(DMODEL * DMODEL) / 256, 256, 0, stream>>>(Wo, wot, DMODEL, DMODEL);

  // 2) QKV projection + RoPE:  3 * 64 * 16 = 3072 wave tiles, 4 waves/block
  qkv_rope_kernel<<<(3 * 64 * NHEAD) / 4, 128, 0, stream>>>(xh, wqt, wkt, wvt, qf, kf, vt);

  // 3) flash attention: 16 heads * 64 q-blocks = 1024 blocks (4 waves each)
  flash_attn_kernel<<<NHEAD * 64, 128, 0, stream>>>(qf, kf, vt, cx);

  // 4) output projection: 64 * 16 = 1024 wave tiles
  out_proj_kernel<<<(64 * NHEAD) / 4, 128, 0, stream>>>(cx, wot, out);
}
